// SpatioTemporalGraphCNN_4337916969721
// MI455X (gfx1250) — matrix operand order repaired and checked by reading, not compiled, not run on hardware
//
#include <hip/hip_runtime.h>

// ---------------------------------------------------------------------------
// SpatioTemporalGraphCNN forward on gfx1250: bf16 WMMA + async global->LDS.
//   support_t = (x @ W12 + bvec)^T per (b,t)    (M=393216, K=64,  N=256)
//   agg       = adj[b] @ support[b,t]           (M=512,   K=512, N=256) x768
//   out       = relu((agg @ w2c^T)*scale+shift) (M=393216, K=256, N=256)
// W12 = w1[:,:,1]^T @ gc_w ; gc_b/b2/BN folded into scale/shift.
// x and adj pre-converted to bf16; all GEMM operands are bf16 [row][K] =>
// staged with global_load_async_to_lds_b128 into double-buffered LDS.
// ---------------------------------------------------------------------------

typedef __attribute__((ext_vector_type(16))) __bf16 v16bf;
typedef __attribute__((ext_vector_type(8)))  __bf16 v8bf;
typedef __attribute__((ext_vector_type(4)))  __bf16 v4bf;
typedef __attribute__((ext_vector_type(8)))  float  v8f;
typedef __attribute__((ext_vector_type(4)))  float  v4f;

#define LDST 40          // padded LDS row stride (bf16): 80B, 16B-aligned
#define BB 32
#define TT 24
#define NNODE 512
#define FF 64
#define HH 256
#define MTOT (BB*TT*NNODE)   // 393216

// ------------------------- prep kernels -----------------------------------

__global__ __launch_bounds__(64) void prep_weights(
    const float* __restrict__ w1,  const float* __restrict__ b1,
    const float* __restrict__ gcw, const float* __restrict__ gcb,
    const float* __restrict__ w2,  const float* __restrict__ b2,
    const float* __restrict__ gamma, const float* __restrict__ beta,
    const float* __restrict__ rmean, const float* __restrict__ rvar,
    __bf16* __restrict__ W12T, float* __restrict__ bvec,
    float* __restrict__ scaleF, float* __restrict__ shiftF)
{
  const int n = blockIdx.x;    // 0..255  (channel k of stage-2 output)
  const int f = threadIdx.x;   // 0..63
  float s = 0.f;
  for (int h = 0; h < HH; ++h)
    s += w1[(h*FF + f)*3 + 1] * gcw[h*HH + n];   // center tap K//2 == 1
  W12T[n*FF + f] = (__bf16)s;                    // [N][K] layout
  if (f == 0) {
    float sb = 0.f;
    for (int h = 0; h < HH; ++h) sb += b1[h] * gcw[h*HH + n];
    bvec[n] = sb;
  }
  if (f == 1) {
    float inv = gamma[n] * rsqrtf(rvar[n] + 1e-5f);
    float extra = b2[n];
    for (int k = 0; k < HH; ++k) extra += gcb[k] * w2[(n*HH + k)*3 + 1];
    scaleF[n] = inv;
    shiftF[n] = extra*inv + beta[n] - rmean[n]*inv;
  }
}

__global__ __launch_bounds__(256) void prep_w2(const float* __restrict__ w2,
                                               __bf16* __restrict__ w2cB)
{
  int i = blockIdx.x*256 + threadIdx.x;  // n*HH + k
  w2cB[i] = (__bf16)w2[i*3 + 1];
}

__global__ __launch_bounds__(256) void cvt_f32_bf16(const float* __restrict__ src,
                                                    __bf16* __restrict__ dst)
{
  const long long i = (long long)blockIdx.x*256 + threadIdx.x;   // x4 elems
  const v4f v = reinterpret_cast<const v4f*>(src)[i];
  v4bf o; o[0]=(__bf16)v.x; o[1]=(__bf16)v.y; o[2]=(__bf16)v.z; o[3]=(__bf16)v.w;
  reinterpret_cast<v4bf*>(dst)[i] = o;
}

// ------------------------- async staging helpers ---------------------------

__device__ __forceinline__ void async_wait0() {
  asm volatile("s_wait_asynccnt 0x0" ::: "memory");
}

// Stage one 128x32 bf16 tile ([row][K], row stride ld) into LDS at lds_base,
// padded row stride LDST, using async global->LDS b128 copies (ASYNCcnt).
__device__ __forceinline__ void stage_tile_async(const __bf16* __restrict__ g,
                                                 unsigned lds_base, int ld,
                                                 int k0, int tid)
{
#pragma unroll
  for (int j = 0; j < 2; ++j) {
    const int q   = tid + 256*j;       // 0..511 chunk id (16B each)
    const int row = q >> 2;            // 0..127
    const int kc  = (q & 3) * 8;       // 0,8,16,24 (bf16 elems)
    const void* ga = (const void*)(g + (long long)row*ld + k0 + kc);
    const unsigned lo = lds_base + (unsigned)(row*LDST + kc)*2u;
    asm volatile("global_load_async_to_lds_b128 %0, %1, off"
                 :: "v"(lo), "v"(ga) : "memory");
  }
}

// ------------------------- tiled WMMA GEMM --------------------------------
// C[128x128 tile] = A[M][K] * B^T where B is [N][K] row-major, all bf16,
// double-buffered async LDS staging, fp32 accumulate.
// EPI: 0 = +bias[col], TRANSPOSED bf16 store (support_t, packed b128)
//      1 = bf16 store (row-major)
//      2 = relu(v*scale[col]+shift[col]), f32 store (row-major)

template<int EPI>
__global__ __launch_bounds__(256) void gemm_wmma(
    const __bf16* __restrict__ Abase, const __bf16* __restrict__ Bbase,
    void* __restrict__ Cb, int Kdim, int ldA, int ldB, int ldC,
    long long aStride, int aDiv, long long bStride, long long cStride,
    const float* __restrict__ p0, const float* __restrict__ p1)
{
  __shared__ __bf16 ldsA[2][128*LDST];
  __shared__ __bf16 ldsB[2][128*LDST];

  const int tid  = threadIdx.x;
  const int z    = blockIdx.z;
  const int n0   = blockIdx.y * 128;

  const __bf16* A = Abase + (long long)(z / aDiv) * aStride
                          + (long long)blockIdx.x * 128 * ldA;
  const __bf16* B = Bbase + (long long)z * bStride + (long long)n0 * ldB;

  const unsigned ldsAo[2] = { (unsigned)(unsigned long long)(&ldsA[0][0]),
                              (unsigned)(unsigned long long)(&ldsA[1][0]) };
  const unsigned ldsBo[2] = { (unsigned)(unsigned long long)(&ldsB[0][0]),
                              (unsigned)(unsigned long long)(&ldsB[1][0]) };

  const int wave  = tid >> 5;
  const int lane  = tid & 31;
  const int waveM = wave >> 2;   // 0..1 -> 64-row half
  const int waveN = wave & 3;    // 0..3 -> 32-col slice
  const int r     = lane & 15;
  const int half  = lane >> 4;

  v8f acc[4][2];
#pragma unroll
  for (int mi = 0; mi < 4; ++mi)
#pragma unroll
    for (int ni = 0; ni < 2; ++ni) acc[mi][ni] = (v8f){0,0,0,0,0,0,0,0};

  const int nk = Kdim >> 5;

  // prologue: stage tile 0 into buffer 0
  stage_tile_async(A, ldsAo[0], ldA, 0, tid);
  stage_tile_async(B, ldsBo[0], ldB, 0, tid);

  for (int s = 0; s < nk; ++s) {
    const int buf = s & 1;
    async_wait0();        // my async copies for tile s are in LDS
    __syncthreads();      // everyone's are; buf^1 free (all read tile s-1)

    if (s + 1 < nk) {     // uniform branch: prefetch next tile, overlaps WMMA
      stage_tile_async(A, ldsAo[buf^1], ldA, (s+1)*32, tid);
      stage_tile_async(B, ldsBo[buf^1], ldB, (s+1)*32, tid);
    }

    const __bf16* la = &ldsA[buf][0];
    const __bf16* lb = &ldsB[buf][0];

    // B fragments (32x16: K = (e&7) + (e>>3)*16 + half*8, N = r)
    v16bf bfrag[2];
#pragma unroll
    for (int ni = 0; ni < 2; ++ni) {
      const int n = waveN*32 + ni*16 + r;
      const v8bf blo = *reinterpret_cast<const v8bf*>(&lb[n*LDST + half*8]);
      const v8bf bhi = *reinterpret_cast<const v8bf*>(&lb[n*LDST + 16 + half*8]);
      bfrag[ni] = __builtin_shufflevector(blo, bhi, 0,1,2,3,4,5,6,7,8,9,10,11,12,13,14,15);
    }
    // A fragments (16x32: K = (e&7) + (e>>3)*16 + half*8, M = r) + WMMA
#pragma unroll
    for (int mi = 0; mi < 4; ++mi) {
      const int row = waveM*64 + mi*16 + r;
      const v8bf alo = *reinterpret_cast<const v8bf*>(&la[row*LDST + half*8]);
      const v8bf ahi = *reinterpret_cast<const v8bf*>(&la[row*LDST + 16 + half*8]);
      const v16bf a = __builtin_shufflevector(alo, ahi, 0,1,2,3,4,5,6,7,8,9,10,11,12,13,14,15);
#pragma unroll
      for (int ni = 0; ni < 2; ++ni) {
        acc[mi][ni] = __builtin_amdgcn_wmma_f32_16x16x32_bf16(
            false, a, false, bfrag[ni], (short)0, acc[mi][ni], false, false);
      }
    }
    __syncthreads();      // done reading buf before its async overwrite lands
  }

  // ---- epilogue: C element (VGPR v, lane) -> M = v + half*8, N = r ----
  if constexpr (EPI == 0) {
    // transposed store: support_t[bt][ch][node], 4 M-tiles per bt block
    const long long base = (long long)(blockIdx.x >> 2) * cStride;
    const int m0l = (blockIdx.x & 3) * 128;
#pragma unroll
    for (int mi = 0; mi < 4; ++mi)
#pragma unroll
      for (int ni = 0; ni < 2; ++ni) {
        const int col = n0 + waveN*32 + ni*16 + r;
        const int m   = m0l + waveM*64 + mi*16 + half*8;
        const float bias = p0[col];
        v8bf pk;
#pragma unroll
        for (int v = 0; v < 8; ++v) pk[v] = (__bf16)(acc[mi][ni][v] + bias);
        *reinterpret_cast<v8bf*>(((__bf16*)Cb) + base + (long long)col*ldC + m) = pk;
      }
  } else {
    const long long crow0 = (long long)z * cStride
                          + (long long)blockIdx.x * 128 * ldC;
#pragma unroll
    for (int mi = 0; mi < 4; ++mi)
#pragma unroll
      for (int ni = 0; ni < 2; ++ni)
#pragma unroll
        for (int v = 0; v < 8; ++v) {
          const int rowL = waveM*64 + mi*16 + half*8 + v;
          const int col  = n0 + waveN*32 + ni*16 + r;
          const long long idx = crow0 + (long long)rowL*ldC + col;
          const float val = acc[mi][ni][v];
          if constexpr (EPI == 1) {
            ((__bf16*)Cb)[idx] = (__bf16)val;
          } else {
            const float o = val * p0[col] + p1[col];
            ((float*)Cb)[idx] = o > 0.f ? o : 0.f;
          }
        }
  }
}

// ------------------------- host launcher ----------------------------------

extern "C" void kernel_launch(void* const* d_in, const int* in_sizes, int n_in,
                              void* d_out, int out_size, void* d_ws, size_t ws_size,
                              hipStream_t stream) {
  const float* x     = (const float*)d_in[0];
  const float* adj   = (const float*)d_in[1];
  const float* w1    = (const float*)d_in[2];
  const float* b1    = (const float*)d_in[3];
  const float* gcw   = (const float*)d_in[4];
  const float* gcb   = (const float*)d_in[5];
  const float* w2    = (const float*)d_in[6];
  const float* b2    = (const float*)d_in[7];
  const float* gamma = (const float*)d_in[8];
  const float* beta  = (const float*)d_in[9];
  const float* rmean = (const float*)d_in[10];
  const float* rvar  = (const float*)d_in[11];

  // workspace layout (bytes): ~470 MB
  char* ws = (char*)d_ws;
  __bf16* W12T     = (__bf16*)(ws + 0);                        // 32768
  float*  bvec     = (float*)(ws + 32768);                     // 1024
  float*  scaleF   = (float*)(ws + 33792);                     // 1024
  float*  shiftF   = (float*)(ws + 34816);                     // 1024
  __bf16* w2cB     = (__bf16*)(ws + 35840);                    // 131072
  __bf16* support_t= (__bf16*)(ws + 166912);                   // 201326592
  __bf16* agg      = (__bf16*)(ws + 166912 + 201326592LL);     // 201326592
  __bf16* x_bf     = (__bf16*)(ws + 402820096LL);              // 50331648
  __bf16* adj_bf   = (__bf16*)(ws + 453151744LL);              // 16777216

  prep_weights<<<dim3(HH), dim3(FF), 0, stream>>>(
      w1, b1, gcw, gcb, w2, b2, gamma, beta, rmean, rvar,
      W12T, bvec, scaleF, shiftF);
  prep_w2<<<dim3(HH*HH/256), dim3(256), 0, stream>>>(w2, w2cB);
  cvt_f32_bf16<<<dim3(MTOT*FF/4/256), dim3(256), 0, stream>>>(x, x_bf);
  cvt_f32_bf16<<<dim3(BB*NNODE*NNODE/4/256), dim3(256), 0, stream>>>(adj, adj_bf);

  // support_t[bt][ch][node] = (x @ W12 + bvec)^T   (bf16, transposed store)
  gemm_wmma<0><<<dim3(MTOT/128, 2, 1), 256, 0, stream>>>(
      x_bf, W12T, (void*)support_t, /*K*/FF, /*ldA*/FF, /*ldB*/FF,
      /*ldC(node stride)*/NNODE, 0LL, 1, 0LL, (long long)HH*NNODE,
      bvec, nullptr);

  // agg[bt][node][ch] = adj[b] @ support[b,t]  (B = support_t is [ch][node])
  gemm_wmma<1><<<dim3(NNODE/128, 2, BB*TT), 256, 0, stream>>>(
      adj_bf, support_t, (void*)agg, /*K*/NNODE, /*ldA*/NNODE, /*ldB*/NNODE,
      /*ldC*/HH, (long long)NNODE*NNODE, TT,
      (long long)HH*NNODE, (long long)NNODE*HH, nullptr, nullptr);

  // out = relu((agg @ w2c^T)*scale + shift)   (f32)
  gemm_wmma<2><<<dim3(MTOT/128, 2, 1), 256, 0, stream>>>(
      agg, w2cB, d_out, /*K*/HH, /*ldA*/HH, /*ldB*/HH, /*ldC*/HH,
      0LL, 1, 0LL, 0LL, scaleF, shiftF);
}
